// CLGNN_2774548873293
// MI455X (gfx1250) — matrix operand order repaired and checked by reading, not compiled, not run on hardware
//
#include <hip/hip_runtime.h>
#include <math.h>

#define EPSv 1e-6f
#define INV_SQRT2 0.70710678118654752440f

typedef __attribute__((ext_vector_type(16))) _Float16 v16h;
typedef __attribute__((ext_vector_type(8)))  _Float16 v8h;
typedef __attribute__((ext_vector_type(8)))  float    v8f;

__constant__ int GRc[8]  = {0,1,1,1,2,2,2,3};   // grade of each blade
__constant__ int BLc[8]  = {0,1,2,4,3,5,6,7};   // blade bitmask per index
__constant__ int IDXc[8] = {0,1,2,4,3,5,6,7};   // bitmask -> blade index (involution)

__device__ inline float sigmoidf_(float x){ return 1.f/(1.f + __expf(-x)); }

// ---------------------------------------------------------------------------
// Fused per-blade channel-mixing linear (WMMA) + optional blade-coupled epilogue.
//   OUT[n,o,i] = sum_m X[n,m,i] * W[o,m,GR[i]]  (+ bias[o] on blade 0)
//   epi = 0: plain (+bias), direct store from accumulators
//   epi = 1: + mv_silu   (aux1 = silu_a, aux2 = silu_b)   via LDS exchange
//   epi = 2: + mv_normalize (aux1 = norm_a)               via LDS exchange
// X:(M,ci,8) f32, W:(co,ci,4) f32, OUT:(M,co,8) f32.  Requires ci<=64, co<=32.
// Block = 256 threads = 8 waves; wave w owns blade w; 16 rows per block.
// ---------------------------------------------------------------------------
__global__ __launch_bounds__(256)
void mv_linear_fused_kernel(const float* __restrict__ X,
                            const float* __restrict__ W,
                            const float* __restrict__ bias,
                            const float* __restrict__ aux1,
                            const float* __restrict__ aux2,
                            float* __restrict__ OUT,
                            int M, int ci, int co, int epi)
{
    __shared__ _Float16 ldsX[8][16][64];   // [blade][row][K]   16 KB
    __shared__ _Float16 ldsW[4][32][64];   // [grade][col][K]   16 KB
    __shared__ float    ldsT[16][32][8];   // [row][col][blade] 16 KB

    const int tid  = threadIdx.x;
    const int wid  = tid >> 5;             // wave id == blade
    const int lane = tid & 31;
    const int hf   = lane >> 4;
    const int l15  = lane & 15;
    const int row0 = blockIdx.x * 16;

    // zero staging (covers K/M padding)
    {
        unsigned* zX = (unsigned*)&ldsX[0][0][0];
        unsigned* zW = (unsigned*)&ldsW[0][0][0];
        for (int i = tid; i < 4096; i += 256){ zX[i] = 0u; zW[i] = 0u; }
    }
    __syncthreads();

    // stage X tile (16 contiguous rows) with coalesced float4 loads
    {
        const int nf = 16*ci*8;
        const float* base = X + (size_t)row0*ci*8;
        for (int f = tid*4; f < nf; f += 1024){
            int r = f / (ci*8);
            if (row0 + r < M){
                const float4 v = *(const float4*)(base + f);
                int rem = f - r*(ci*8);
                int k  = rem >> 3;
                int b0 = rem & 7;          // 0 or 4
                ldsX[b0+0][r][k] = (_Float16)v.x;
                ldsX[b0+1][r][k] = (_Float16)v.y;
                ldsX[b0+2][r][k] = (_Float16)v.z;
                ldsX[b0+3][r][k] = (_Float16)v.w;
            }
        }
    }
    // stage W (co,ci,4): each float4 = the 4 grades of one (o,k)
    {
        const int nf = co*ci*4;
        for (int f = tid*4; f < nf; f += 1024){
            const float4 v = *(const float4*)(W + f);
            int o = f / (ci*4);
            int k = (f - o*(ci*4)) >> 2;
            ldsW[0][o][k] = (_Float16)v.x;
            ldsW[1][o][k] = (_Float16)v.y;
            ldsW[2][o][k] = (_Float16)v.z;
            ldsW[3][o][k] = (_Float16)v.w;
        }
    }
    __syncthreads();

    const int blade  = wid;
    const int g      = GRc[blade];
    const int ksteps = (ci + 31) / 32;
    const int ntiles = (co + 15) / 16;

    v8f acc0 = {}; v8f acc1 = {};
    for (int ks = 0; ks < ksteps; ++ks){
        const int kb = ks * 32;
        // A fragment: row = l15; K = kb + 16h + 0..15 (contiguous)
        const _Float16* ax = &ldsX[blade][l15][0];
        v8h alo = *(const v8h*)(ax + kb + 16*hf);
        v8h ahi = *(const v8h*)(ax + kb + 16*hf + 8);
        v16h a;
#pragma unroll
        for (int t = 0; t < 8; ++t){ a[t] = alo[t]; a[8+t] = ahi[t]; }
        // B fragments: col = nt*16 + l15; K = kb + 16h + 0..15 (contiguous)
        {
            const _Float16* bx = &ldsW[g][l15][0] + kb + 16*hf;
            v8h blo = *(const v8h*)(bx);
            v8h bhi = *(const v8h*)(bx + 8);
            v16h b;
#pragma unroll
            for (int t = 0; t < 8; ++t){ b[t] = blo[t]; b[8+t] = bhi[t]; }
            acc0 = __builtin_amdgcn_wmma_f32_16x16x32_f16(false, a, false, b,
                                                          (short)0, acc0, false, false);
        }
        if (ntiles > 1){   // wave-uniform branch: EXEC stays all-ones
            const _Float16* bx = &ldsW[g][16 + l15][0] + kb + 16*hf;
            v8h blo = *(const v8h*)(bx);
            v8h bhi = *(const v8h*)(bx + 8);
            v16h b;
#pragma unroll
            for (int t = 0; t < 8; ++t){ b[t] = blo[t]; b[8+t] = bhi[t]; }
            acc1 = __builtin_amdgcn_wmma_f32_16x16x32_f16(false, a, false, b,
                                                          (short)0, acc1, false, false);
        }
    }

    if (epi == 0){
        // direct store with optional bias on blade 0 (C/D: VGPR r -> M=r+8h, N=l15)
#pragma unroll
        for (int nt = 0; nt < 2; ++nt){
            if (nt >= ntiles) break;
            int o = nt*16 + l15;
#pragma unroll
            for (int r = 0; r < 8; ++r){
                int row = row0 + r + 8*hf;
                if (row < M && o < co){
                    float v = (nt == 0) ? acc0[r] : acc1[r];
                    if (blade == 0 && bias) v += bias[o];
                    OUT[((size_t)row*co + o)*8 + blade] = v;
                }
            }
        }
        return;
    }

    // blade-coupled epilogue: exchange C through LDS
#pragma unroll
    for (int nt = 0; nt < 2; ++nt){
        if (nt >= ntiles) break;
        int col = nt*16 + l15;
        float bv = (blade == 0 && bias && col < co) ? bias[col] : 0.f;
#pragma unroll
        for (int r = 0; r < 8; ++r)
            ldsT[r + 8*hf][col][blade] = ((nt == 0) ? acc0[r] : acc1[r]) + bv;
    }
    __syncthreads();

    const int tot = 16*co;
    for (int e = tid; e < tot; e += 256){
        int r = e / co, c = e - r*co;
        if (row0 + r >= M) continue;
        float v[8];
#pragma unroll
        for (int i = 0; i < 8; ++i) v[i] = ldsT[r][c][i];
        if (epi == 1){                      // mv_silu
            float inv[4];
            inv[0] = v[0];
            inv[1] = v[1]*v[1] + v[2]*v[2] + v[3]*v[3];
            inv[2] = v[4]*v[4] + v[5]*v[5] + v[6]*v[6];
            inv[3] = v[7]*v[7];
            float s[4];
#pragma unroll
            for (int gg = 0; gg < 4; ++gg)
                s[gg] = sigmoidf_(aux1[c*4+gg]*inv[gg] + aux2[c*4+gg]);
#pragma unroll
            for (int i = 0; i < 8; ++i) v[i] *= s[GRc[i]];
        } else {                            // mv_normalize
            float gn[4];
            gn[0] = fabsf(v[0]);
            gn[1] = sqrtf(v[1]*v[1] + v[2]*v[2] + v[3]*v[3]);
            gn[2] = sqrtf(v[4]*v[4] + v[5]*v[5] + v[6]*v[6]);
            gn[3] = fabsf(v[7]);
            float nm[4];
#pragma unroll
            for (int gg = 0; gg < 4; ++gg){
                float s = sigmoidf_(aux1[c*4+gg]);
                nm[gg] = s*(gn[gg] - 1.f) + 1.f;
            }
#pragma unroll
            for (int i = 0; i < 8; ++i) v[i] /= (nm[GRc[i]] + EPSv);
        }
        float* op = OUT + ((size_t)(row0 + r)*co + c)*8;
#pragma unroll
        for (int i = 0; i < 8; ++i) op[i] = v[i];
    }
}

// ---------------------------------------------------------------------------
// Pointwise / reduction kernels
// ---------------------------------------------------------------------------
__global__ void zero_kernel(float* p, long long n){
    long long i = (long long)blockIdx.x*blockDim.x + threadIdx.x;
    if (i < n) p[i] = 0.f;
}

__global__ void mean_kernel(const float* __restrict__ loc, float* __restrict__ meanB,
                            int npb){
    __shared__ float sm[3][64];
    int b = blockIdx.x, t = threadIdx.x;
    float s0=0.f, s1=0.f, s2=0.f;
    for (int i = t; i < npb; i += 64){
        const float* p = loc + (size_t)(b*npb + i)*3;
        s0 += p[0]; s1 += p[1]; s2 += p[2];
    }
    sm[0][t]=s0; sm[1][t]=s1; sm[2][t]=s2;
    __syncthreads();
    for (int w = 32; w > 0; w >>= 1){
        if (t < w){ sm[0][t]+=sm[0][t+w]; sm[1][t]+=sm[1][t+w]; sm[2][t]+=sm[2][t+w]; }
        __syncthreads();
    }
    if (t == 0){
        meanB[b*3+0]=sm[0][0]/npb; meanB[b*3+1]=sm[1][0]/npb; meanB[b*3+2]=sm[2][0]/npb;
    }
}

__global__ void embed_kernel(const float* __restrict__ charges, const float* __restrict__ loc,
                             const float* __restrict__ vel, const float* __restrict__ meanB,
                             const float* __restrict__ ew, const float* __restrict__ eb,
                             float* __restrict__ h, int N, int H, int npb){
    int idx = blockIdx.x*blockDim.x + threadIdx.x;
    if (idx >= N*H) return;
    int n = idx / H, o = idx % H;
    int b = n / npb;
    float lm0 = loc[n*3+0]-meanB[b*3+0];
    float lm1 = loc[n*3+1]-meanB[b*3+1];
    float lm2 = loc[n*3+2]-meanB[b*3+2];
    const float* w = ew + o*3;
    float* out = h + ((size_t)n*H + o)*8;
    out[0] = charges[n]*w[0] + eb[o];
    out[1] = lm0*w[1] + vel[n*3+0]*w[2];
    out[2] = lm1*w[1] + vel[n*3+1]*w[2];
    out[3] = lm2*w[1] + vel[n*3+2]*w[2];
    out[4]=0.f; out[5]=0.f; out[6]=0.f; out[7]=0.f;
}

__global__ void edge_input_kernel(const float* __restrict__ h, const float* __restrict__ ea,
                                  const int* __restrict__ src, const int* __restrict__ dst,
                                  float* __restrict__ X, int E, int H){
    int idx = blockIdx.x*blockDim.x + threadIdx.x;
    int ci = H + 1;
    if (idx >= E*ci) return;
    int e = idx / ci, c = idx % ci;
    float* out = X + ((size_t)e*ci + c)*8;
    if (c < H){
        const float* hd = h + ((size_t)dst[e]*H + c)*8;
        const float* hs = h + ((size_t)src[e]*H + c)*8;
#pragma unroll
        for (int i = 0; i < 8; ++i) out[i] = hd[i] - hs[i];
    } else {
        out[0] = ea[e];
#pragma unroll
        for (int i = 1; i < 8; ++i) out[i] = 0.f;
    }
}

// O (holding left-linear result) <- (O + geom(T, R)) / sqrt(2)
__global__ void geom_kernel(const float* __restrict__ T, const float* __restrict__ R,
                            float* __restrict__ O, const float* __restrict__ gwAll,
                            int M, int co){
    int idx = blockIdx.x*blockDim.x + threadIdx.x;
    if (idx >= M*co) return;
    int c = idx % co;
    const float* t = T + (size_t)idx*8;
    const float* r = R + (size_t)idx*8;
    float*       o = O + (size_t)idx*8;
    const float* gw = gwAll + (size_t)c*64;
    float tv[8], rv[8];
#pragma unroll
    for (int i = 0; i < 8; ++i){ tv[i]=t[i]; rv[i]=r[i]; }
#pragma unroll
    for (int j = 0; j < 8; ++j){
        int bj = BLc[j], gj = GRc[j];
        float acc = 0.f;
#pragma unroll
        for (int i = 0; i < 8; ++i){
            int ai = BLc[i];
            int k  = IDXc[ai ^ bj];
            int s = 0, aa = ai >> 1;
            while (aa){ s += __popc(aa & bj); aa >>= 1; }
            float sgn = (s & 1) ? -1.f : 1.f;
            acc += tv[i] * sgn * gw[GRc[i]*16 + gj*4 + GRc[k]] * rv[k];
        }
        o[j] = (o[j] + acc) * INV_SQRT2;
    }
}

__global__ void layernorm_kernel(float* __restrict__ O, const float* __restrict__ la,
                                 int M, int co){
    int n = blockIdx.x*blockDim.x + threadIdx.x;
    if (n >= M) return;
    float* base = O + (size_t)n*co*8;
    float sum = 0.f;
    for (int c = 0; c < co; ++c){
        float s2 = 0.f;
#pragma unroll
        for (int i = 0; i < 8; ++i){ float v = base[c*8+i]; s2 += v*v; }
        sum += sqrtf(s2);
    }
    float mean = sum/(float)co + EPSv;
    for (int c = 0; c < co; ++c){
        float scale = la[c]/mean;
#pragma unroll
        for (int i = 0; i < 8; ++i) base[c*8+i] *= scale;
    }
}

__global__ void scatter_kernel(const float* __restrict__ msg, const int* __restrict__ dst,
                               float* __restrict__ agg, float* __restrict__ cnt,
                               int E, int H){
    int idx = blockIdx.x*blockDim.x + threadIdx.x;
    if (idx >= E*H) return;
    int e = idx / H, c = idx % H;
    int d = dst[e];
    const float* m = msg + (size_t)idx*8;
    float* a = agg + ((size_t)d*H + c)*8;
#pragma unroll
    for (int i = 0; i < 8; ++i) atomicAdd(&a[i], m[i]);
    if (c == 0) atomicAdd(&cnt[d], 1.f);
}

__global__ void agg_div_kernel(float* __restrict__ agg, const float* __restrict__ cnt,
                               int N, int H){
    int idx = blockIdx.x*blockDim.x + threadIdx.x;
    if (idx >= N*H) return;
    float den = fmaxf(cnt[idx / H], 1.f);
    float* a = agg + (size_t)idx*8;
#pragma unroll
    for (int i = 0; i < 8; ++i) a[i] /= den;
}

__global__ void node_input_kernel(const float* __restrict__ h, const float* __restrict__ agg,
                                  float* __restrict__ X, int N, int H){
    int idx = blockIdx.x*blockDim.x + threadIdx.x;
    int ci = 2*H;
    if (idx >= N*ci) return;
    int n = idx / ci, c = idx % ci;
    const float* s = (c < H) ? h   + ((size_t)n*H + c    )*8
                             : agg + ((size_t)n*H + c - H)*8;
    float* o = X + ((size_t)n*ci + c)*8;
#pragma unroll
    for (int i = 0; i < 8; ++i) o[i] = s[i];
}

__global__ void h_add_kernel(float* __restrict__ h, const float* __restrict__ d, long long n){
    long long i = (long long)blockIdx.x*blockDim.x + threadIdx.x;
    if (i < n) h[i] += d[i];
}

__global__ void loss_kernel(const float* __restrict__ loc, const float* __restrict__ y,
                            const float* __restrict__ hp, float* __restrict__ scal, int N){
    int idx = blockIdx.x*blockDim.x + threadIdx.x;
    if (idx >= N*3) return;
    int n = idx / 3, d = idx % 3;
    float yv = y[idx];
    float pred = loc[idx] + hp[(size_t)n*8 + 1 + d];
    if (yv == yv){                    // ~isnan
        float df = pred - yv;
        atomicAdd(&scal[0], df*df);
        atomicAdd(&scal[1], 1.f);
    }
}

__global__ void finish_kernel(const float* __restrict__ scal, float* __restrict__ out){
    out[0] = scal[0] / fmaxf(scal[1], 1.f);
}

// ---------------------------------------------------------------------------
// Host orchestration
// ---------------------------------------------------------------------------
struct SubP {
    const float *geom_w,*left_b,*left_w,*norm_a,*right_w,*lin_b,*lin_w,*ln_a,*silu_a,*silu_b;
    int ci, co;
};

struct LeafSrc {
    void* const* d_in; int idx;
    const float* flat; size_t off;
    bool array_mode;
    const float* next(size_t cnt){
        if (array_mode) return (const float*)d_in[idx++];
        const float* p = flat + off; off += cnt; return p;
    }
};

static void fill_params(LeafSrc& s, int H,
                        const float*& embed_b, const float*& embed_w,
                        SubP sub[3][4], const float*& proj_b, const float*& proj_w){
    embed_b = s.next((size_t)H);
    embed_w = s.next((size_t)H*3);
    int cis[4] = {H+1, H, 2*H, H};
    for (int L = 0; L < 3; ++L)
        for (int k = 0; k < 4; ++k){
            SubP& p = sub[L][k];
            p.ci = cis[k]; p.co = H;
            p.geom_w = s.next((size_t)H*64);
            p.left_b = s.next((size_t)H);
            p.left_w = s.next((size_t)H*H*4);
            p.norm_a = s.next((size_t)H*4);
            p.right_w= s.next((size_t)H*H*4);
            p.lin_b  = s.next((size_t)H);
            p.lin_w  = s.next((size_t)H*p.ci*4);
            p.ln_a   = s.next((size_t)H);
            p.silu_a = s.next((size_t)H*4);
            p.silu_b = s.next((size_t)H*4);
        }
    proj_b = s.next(1);
    proj_w = s.next((size_t)H*4);
}

static inline int nblk(long long n, int b){ return (int)((n + b - 1)/b); }

static void run_sub(const SubP& p, const float* Xin, float* T, float* R, float* O,
                    int M, hipStream_t stream){
    dim3 blk(256);
    dim3 grd((M+15)/16);
    // lin + bias + silu (fused)
    mv_linear_fused_kernel<<<grd, blk, 0, stream>>>(Xin, p.lin_w, p.lin_b,
                                                    p.silu_a, p.silu_b, T, M, p.ci, p.co, 1);
    // right + normalize (fused)
    mv_linear_fused_kernel<<<grd, blk, 0, stream>>>(T, p.right_w, nullptr,
                                                    p.norm_a, nullptr, R, M, p.co, p.co, 2);
    // left + bias (plain)
    mv_linear_fused_kernel<<<grd, blk, 0, stream>>>(T, p.left_w, p.left_b,
                                                    nullptr, nullptr, O, M, p.co, p.co, 0);
    geom_kernel<<<nblk((long long)M*p.co,256),256,0,stream>>>(T, R, O, p.geom_w, M, p.co);
    layernorm_kernel<<<nblk(M,256),256,0,stream>>>(O, p.ln_a, M, p.co);
}

extern "C" void kernel_launch(void* const* d_in, const int* in_sizes, int n_in,
                              void* d_out, int out_size, void* d_ws, size_t ws_size,
                              hipStream_t stream){
    (void)out_size; (void)ws_size;
    const float *charges=nullptr, *edge_attr=nullptr, *loc=nullptr, *vel=nullptr, *y=nullptr;
    const int *eidx=nullptr;
    const float *embed_b=nullptr, *embed_w=nullptr, *proj_b=nullptr, *proj_w=nullptr;
    SubP sub[3][4];
    int N=0, E=0, B=1, H=28;

    if (n_in >= 100){
        if (in_sizes[0] > 1000){
            // full sorted tree_leaves: charges, edge_attr, edge_index, loc,
            //   [124 param leaves @4..127], ptr, vel, y
            charges   = (const float*)d_in[0];
            edge_attr = (const float*)d_in[1];
            eidx      = (const int*)  d_in[2];
            loc       = (const float*)d_in[3];
            vel       = (const float*)d_in[129];
            y         = (const float*)d_in[130];
            N = in_sizes[0]; E = in_sizes[1]; B = in_sizes[128]-1; H = in_sizes[4];
            LeafSrc s{d_in, 4, nullptr, 0, true};
            fill_params(s, H, embed_b, embed_w, sub, proj_b, proj_w);
        } else {
            // params-first: [124 param leaves @0..123], loc, vel, charges, y,
            //   edge_attr, edge_index, ptr
            loc       = (const float*)d_in[124];
            vel       = (const float*)d_in[125];
            charges   = (const float*)d_in[126];
            y         = (const float*)d_in[127];
            edge_attr = (const float*)d_in[128];
            eidx      = (const int*)  d_in[129];
            N = in_sizes[126]; E = in_sizes[128]; B = in_sizes[130]-1; H = in_sizes[0];
            LeafSrc s{d_in, 0, nullptr, 0, true};
            fill_params(s, H, embed_b, embed_w, sub, proj_b, proj_w);
        }
    } else {
        // 8 inputs, insertion order; params concatenated flat (sorted-leaf order)
        loc       = (const float*)d_in[1];
        vel       = (const float*)d_in[2];
        charges   = (const float*)d_in[3];
        y         = (const float*)d_in[4];
        edge_attr = (const float*)d_in[5];
        eidx      = (const int*)  d_in[6];
        N = in_sizes[3]; E = in_sizes[5]; B = in_sizes[7]-1; H = 28;
        LeafSrc s{nullptr, 0, (const float*)d_in[0], 0, false};
        fill_params(s, H, embed_b, embed_w, sub, proj_b, proj_w);
    }
    if (B < 1) B = 1;
    const int* srcI = eidx;
    const int* dstI = eidx + E;
    int npb = N / B; if (npb < 1) npb = 1;

    // workspace layout (floats)
    float* ws = (float*)d_ws;
    size_t off = 0;
    auto wsa = [&](size_t n){ float* p = ws + off; off += n; return p; };
    float* h     = wsa((size_t)N*H*8);
    float* agg   = wsa((size_t)N*H*8);
    float* cnt   = wsa((size_t)N);
    float* bufA  = wsa((size_t)E*(H+1)*8);   // edge/node CEMLP input & ping-pong
    float* bufT  = wsa((size_t)E*H*8);
    float* bufR  = wsa((size_t)E*H*8);
    float* bufO  = wsa((size_t)E*H*8);
    float* meanB = wsa((size_t)B*3);
    float* scal  = wsa(4);

    // embedding
    mean_kernel<<<B, 64, 0, stream>>>(loc, meanB, npb);
    embed_kernel<<<nblk((long long)N*H,256),256,0,stream>>>(charges, loc, vel, meanB,
                                                            embed_w, embed_b, h, N, H, npb);
    // 3 EGCL layers
    for (int L = 0; L < 3; ++L){
        edge_input_kernel<<<nblk((long long)E*(H+1),256),256,0,stream>>>(h, edge_attr,
                                                                         srcI, dstI, bufA, E, H);
        run_sub(sub[L][0], bufA, bufT, bufR, bufO, E, stream);   // 29 -> 28
        run_sub(sub[L][1], bufO, bufT, bufR, bufA, E, stream);   // 28 -> 28, msg in bufA
        zero_kernel<<<nblk((long long)N*H*8,256),256,0,stream>>>(agg, (long long)N*H*8);
        zero_kernel<<<nblk(N,256),256,0,stream>>>(cnt, (long long)N);
        scatter_kernel<<<nblk((long long)E*H,256),256,0,stream>>>(bufA, dstI, agg, cnt, E, H);
        agg_div_kernel<<<nblk((long long)N*H,256),256,0,stream>>>(agg, cnt, N, H);
        node_input_kernel<<<nblk((long long)N*2*H,256),256,0,stream>>>(h, agg, bufO, N, H);
        run_sub(sub[L][2], bufO, bufT, bufR, bufA, N, stream);   // 56 -> 28
        run_sub(sub[L][3], bufA, bufT, bufR, bufO, N, stream);   // 28 -> 28
        h_add_kernel<<<nblk((long long)N*H*8,256),256,0,stream>>>(h, bufO, (long long)N*H*8);
    }
    // projection (co = 1) + loss
    mv_linear_fused_kernel<<<dim3((N+15)/16), 256, 0, stream>>>(h, proj_w, proj_b,
                                                                nullptr, nullptr, bufR, N, H, 1, 0);
    zero_kernel<<<1, 32, 0, stream>>>(scal, 4);
    loss_kernel<<<nblk((long long)N*3,256),256,0,stream>>>(loc, y, bufR, scal, N);
    finish_kernel<<<1, 1, 0, stream>>>(scal, (float*)d_out);
}
